// MHA_19121194401830
// MI455X (gfx1250) — matrix operand order repaired and checked by reading, not compiled, not run on hardware
//
#include <hip/hip_runtime.h>
#include <hip/hip_bf16.h>

typedef __attribute__((ext_vector_type(16))) __bf16 v16bf;
typedef __attribute__((ext_vector_type(8)))  __bf16 v8bf;
typedef __attribute__((ext_vector_type(8)))  float  v8f;

#define HIDDEN 1024
#define NHEADS 16
#define DHEAD  64
#define SEQ    2048
#define BATCH  2
#define MTOT   (BATCH * SEQ)   /* 4096 rows of x */

// ---------------------------------------------------------------------------
// WMMA helpers (CDNA5 bf16 16x16x32, f32 accumulate)
// ---------------------------------------------------------------------------
static __device__ __forceinline__ v8f wmma_bf16(v16bf a, v16bf b, v8f c) {
#if defined(__gfx1250__)
  // (neg_a, A, neg_b, B, c_mod, C, reuse_a, reuse_b)
  return __builtin_amdgcn_wmma_f32_16x16x32_bf16(false, a, false, b, (short)0, c,
                                                 false, false);
#else
  (void)a; (void)b;
  return c;
#endif
}

static __device__ __forceinline__ v16bf cat8(v8bf lo, v8bf hi) {
  return __builtin_shufflevector(lo, hi, 0, 1, 2, 3, 4, 5, 6, 7,
                                         8, 9, 10, 11, 12, 13, 14, 15);
}

// A fragment: 16x32 bf16 tile at (row0, k0) of a row-major [*, ld] matrix.
// ISA layout: lanes 0-15 hold M=lane, elems 0-7 -> K=kh*8+[0,8),
// elems 8-15 -> K=16+kh*8+[0,8)  (kh = lane>>4).
static __device__ __forceinline__ v16bf load_A(const __bf16* p, int ld,
                                               int row0, int k0, int lane) {
  int row = row0 + (lane & 15);
  int kh  = lane >> 4;
  const __bf16* base = p + (size_t)row * ld + k0;
  v8bf lo = *(const v8bf*)(base + kh * 8);
  v8bf hi = *(const v8bf*)(base + 16 + kh * 8);
  return cat8(lo, hi);
}

// B fragment: 32x16 bf16 tile at (k0, n0) where column n is contiguous in
// memory at p + n*ld + k.  ISA layout: lane holds column lane&15, elems 0-7 ->
// K = kh*8+[0,8), elems 8-15 -> K = 16+kh*8+[0,8).
static __device__ __forceinline__ v16bf load_B(const __bf16* p, int ld,
                                               int k0, int n0, int lane) {
  int col = n0 + (lane & 15);
  int kh  = lane >> 4;
  const __bf16* base = p + (size_t)col * ld + k0;
  v8bf lo = *(const v8bf*)(base + kh * 8);
  v8bf hi = *(const v8bf*)(base + 16 + kh * 8);
  return cat8(lo, hi);
}

// ---------------------------------------------------------------------------
// fp32 -> bf16 conversion
// ---------------------------------------------------------------------------
__global__ void cvt_f32_bf16(const float* __restrict__ src,
                             __bf16* __restrict__ dst, int n) {
  int i = blockIdx.x * blockDim.x + threadIdx.x;
  if (i < n) dst[i] = (__bf16)src[i];
}

// ---------------------------------------------------------------------------
// Q/K/V projection: y = x @ W^T + bias, written as bf16 with head permute.
// transpose_out==0 -> dst[(b*H + h)*SEQ*D + s*D + d]      (Q, K)
// transpose_out==1 -> dst[(b*H + h)*D*SEQ + d*SEQ + s]    (V transposed)
// Block = 128 threads (4 waves); wave tile = 32 rows x 64 cols.
// ---------------------------------------------------------------------------
__global__ __launch_bounds__(128) void proj_qkv(
    const __bf16* __restrict__ xb,   // [4096, 1024]
    const __bf16* __restrict__ wb,   // [1024(out), 1024(in)]
    const float*  __restrict__ bias, // [1024]
    __bf16* __restrict__ dst, int transpose_out) {
  int lane = threadIdx.x & 31;
  int w    = threadIdx.x >> 5;
  int m0   = blockIdx.x * 128 + w * 32;
  int n0   = blockIdx.y * 64;

  v8f acc[2][4];
#pragma unroll
  for (int i = 0; i < 2; ++i)
#pragma unroll
    for (int nd = 0; nd < 4; ++nd) acc[i][nd] = (v8f)0.0f;

  for (int k0 = 0; k0 < HIDDEN; k0 += 32) {
    v16bf a0 = load_A(xb, HIDDEN, m0,      k0, lane);
    v16bf a1 = load_A(xb, HIDDEN, m0 + 16, k0, lane);
#pragma unroll
    for (int nd = 0; nd < 4; ++nd) {
      v16bf b = load_B(wb, HIDDEN, k0, n0 + nd * 16, lane);
      acc[0][nd] = wmma_bf16(a0, b, acc[0][nd]);
      acc[1][nd] = wmma_bf16(a1, b, acc[1][nd]);
    }
  }

  int kh = lane >> 4;
#pragma unroll
  for (int nd = 0; nd < 4; ++nd) {
    int n = n0 + nd * 16 + (lane & 15);
    float bv = bias[n];
    int h = n >> 6, d = n & 63;
#pragma unroll
    for (int i = 0; i < 2; ++i) {
#pragma unroll
      for (int r = 0; r < 8; ++r) {
        int m = m0 + i * 16 + r + 8 * kh;   // global row in [0,4096)
        int b_ = m >> 11, s = m & (SEQ - 1);
        float v = acc[i][nd][r] + bv;
        size_t idx;
        if (transpose_out)
          idx = (((size_t)(b_ * NHEADS + h) * DHEAD + d) * SEQ) + s;
        else
          idx = (((size_t)(b_ * NHEADS + h) * SEQ + s) * DHEAD) + d;
        dst[idx] = (__bf16)v;
      }
    }
  }
}

// ---------------------------------------------------------------------------
// Flash attention per (b,h): scores = QK^T / 8, online softmax, O = P V.
// Block = 256 threads (8 waves); each wave owns 16 query rows; j-tile = 32.
// Output written bf16 as [b, s, h*64+d]  (i.e. row-major [4096, 1024]).
// ---------------------------------------------------------------------------
__global__ __launch_bounds__(256) void flash_attn(
    const __bf16* __restrict__ qb,   // [32, 2048, 64]
    const __bf16* __restrict__ kb,   // [32, 2048, 64]
    const __bf16* __restrict__ vtb,  // [32, 64, 2048]  (transposed V)
    __bf16* __restrict__ attnb) {    // [4096, 1024]
  __shared__ __align__(32) __bf16 plds[8 * 16 * 32];   // 8 KB, 1 KB per wave

  int lane = threadIdx.x & 31;
  int w    = threadIdx.x >> 5;
  int bh   = blockIdx.x;                 // 0..31
  int b    = bh >> 4, h = bh & 15;
  int q0   = blockIdx.y * 128 + w * 16;  // query row base
  int kh   = lane >> 4;
  int cl   = lane & 15;

  const __bf16* qptr = qb  + (size_t)bh * SEQ * DHEAD;
  const __bf16* kptr = kb  + (size_t)bh * SEQ * DHEAD;
  const __bf16* vptr = vtb + (size_t)bh * DHEAD * SEQ;
  __bf16* pl = plds + w * (16 * 32);

  // Q fragments for this wave's 16 rows (K = d, two halves of 32)
  v16bf qa0 = load_A(qptr, DHEAD, q0, 0,  lane);
  v16bf qa1 = load_A(qptr, DHEAD, q0, 32, lane);

  v8f o[4];
#pragma unroll
  for (int nd = 0; nd < 4; ++nd) o[nd] = (v8f)0.0f;
  float mrow[8], lrow[8];
#pragma unroll
  for (int r = 0; r < 8; ++r) { mrow[r] = -1e30f; lrow[r] = 0.0f; }

  const float scale = 0.125f;  // 1/sqrt(64)

  for (int j0 = 0; j0 < SEQ; j0 += 32) {
    // ---- S = Q K^T for columns [j0, j0+32) ----
    v16bf kb0 = load_B(kptr, DHEAD, 0,  j0,      lane);
    v16bf kb1 = load_B(kptr, DHEAD, 32, j0,      lane);
    v16bf kb2 = load_B(kptr, DHEAD, 0,  j0 + 16, lane);
    v16bf kb3 = load_B(kptr, DHEAD, 32, j0 + 16, lane);
    v8f s0 = (v8f)0.0f, s1 = (v8f)0.0f;
    s0 = wmma_bf16(qa0, kb0, s0);
    s0 = wmma_bf16(qa1, kb1, s0);
    s1 = wmma_bf16(qa0, kb2, s1);
    s1 = wmma_bf16(qa1, kb3, s1);

    // ---- online softmax; stage P into LDS (C-layout -> A-layout) ----
#pragma unroll
    for (int r = 0; r < 8; ++r) {
      float a = s0[r] * scale, c = s1[r] * scale;
      float tm = fmaxf(a, c);
#pragma unroll
      for (int m = 1; m < 16; m <<= 1) tm = fmaxf(tm, __shfl_xor(tm, m, 32));
      float mo = mrow[r], mn = fmaxf(mo, tm);
      float alpha = __expf(mo - mn);
      float p0 = __expf(a - mn), p1 = __expf(c - mn);
      float rs = p0 + p1;
#pragma unroll
      for (int m = 1; m < 16; m <<= 1) rs += __shfl_xor(rs, m, 32);
      lrow[r] = lrow[r] * alpha + rs;
      mrow[r] = mn;
#pragma unroll
      for (int nd = 0; nd < 4; ++nd) o[nd][r] *= alpha;
      int rloc = r + 8 * kh;
      pl[rloc * 32 + cl]      = (__bf16)p0;
      pl[rloc * 32 + cl + 16] = (__bf16)p1;
    }

    // ---- O += P V  (A = P from LDS, K = j over 32) ----
    {
      const __bf16* base = pl + cl * 32;          // row = lane&15
      v8bf lo = *(const v8bf*)(base + kh * 8);
      v8bf hi = *(const v8bf*)(base + 16 + kh * 8);
      v16bf pa = cat8(lo, hi);
#pragma unroll
      for (int nd = 0; nd < 4; ++nd) {
        v16bf vb = load_B(vptr, SEQ, j0, nd * 16, lane);
        o[nd] = wmma_bf16(pa, vb, o[nd]);
      }
    }
  }

  // ---- normalize and store bf16 to [b, s, h*64+d] ----
#pragma unroll
  for (int r = 0; r < 8; ++r) {
    float inv = 1.0f / lrow[r];
    int s = q0 + r + 8 * kh;
    size_t rowbase = ((size_t)b * SEQ + s) * HIDDEN + h * DHEAD;
#pragma unroll
    for (int nd = 0; nd < 4; ++nd)
      attnb[rowbase + nd * 16 + cl] = (__bf16)(o[nd][r] * inv);
  }
}

// ---------------------------------------------------------------------------
// Output projection: out = attn @ Wo^T + bo   (fp32 output)
// ---------------------------------------------------------------------------
__global__ __launch_bounds__(128) void proj_out(
    const __bf16* __restrict__ ab,   // [4096, 1024]
    const __bf16* __restrict__ wb,   // [1024, 1024]
    const float*  __restrict__ bias,
    float* __restrict__ dst) {
  int lane = threadIdx.x & 31;
  int w    = threadIdx.x >> 5;
  int m0   = blockIdx.x * 128 + w * 32;
  int n0   = blockIdx.y * 64;

  v8f acc[2][4];
#pragma unroll
  for (int i = 0; i < 2; ++i)
#pragma unroll
    for (int nd = 0; nd < 4; ++nd) acc[i][nd] = (v8f)0.0f;

  for (int k0 = 0; k0 < HIDDEN; k0 += 32) {
    v16bf a0 = load_A(ab, HIDDEN, m0,      k0, lane);
    v16bf a1 = load_A(ab, HIDDEN, m0 + 16, k0, lane);
#pragma unroll
    for (int nd = 0; nd < 4; ++nd) {
      v16bf b = load_B(wb, HIDDEN, k0, n0 + nd * 16, lane);
      acc[0][nd] = wmma_bf16(a0, b, acc[0][nd]);
      acc[1][nd] = wmma_bf16(a1, b, acc[1][nd]);
    }
  }

  int kh = lane >> 4;
#pragma unroll
  for (int nd = 0; nd < 4; ++nd) {
    int n = n0 + nd * 16 + (lane & 15);
    float bv = bias[n];
#pragma unroll
    for (int i = 0; i < 2; ++i) {
#pragma unroll
      for (int r = 0; r < 8; ++r) {
        int m = m0 + i * 16 + r + 8 * kh;
        dst[(size_t)m * HIDDEN + n] = acc[i][nd][r] + bv;
      }
    }
  }
}

// ---------------------------------------------------------------------------
// Host launch
// ---------------------------------------------------------------------------
extern "C" void kernel_launch(void* const* d_in, const int* in_sizes, int n_in,
                              void* d_out, int out_size, void* d_ws,
                              size_t ws_size, hipStream_t stream) {
  (void)in_sizes; (void)n_in; (void)out_size; (void)ws_size;
  const float* x  = (const float*)d_in[0];
  const float* Wq = (const float*)d_in[1];
  const float* bq = (const float*)d_in[2];
  const float* Wk = (const float*)d_in[3];
  const float* bk = (const float*)d_in[4];
  const float* Wv = (const float*)d_in[5];
  const float* bv = (const float*)d_in[6];
  const float* Wo = (const float*)d_in[7];
  const float* bo = (const float*)d_in[8];
  float* out = (float*)d_out;

  char* ws = (char*)d_ws;
  size_t off = 0;
  auto carve = [&](size_t bytes) -> char* {
    char* p = ws + off;
    off += (bytes + 255) & ~(size_t)255;
    return p;
  };
  __bf16* xb    = (__bf16*)carve((size_t)MTOT * HIDDEN * 2);     // 8 MB
  __bf16* wqb   = (__bf16*)carve((size_t)HIDDEN * HIDDEN * 2);   // 2 MB
  __bf16* wkb   = (__bf16*)carve((size_t)HIDDEN * HIDDEN * 2);
  __bf16* wvb   = (__bf16*)carve((size_t)HIDDEN * HIDDEN * 2);
  __bf16* wob   = (__bf16*)carve((size_t)HIDDEN * HIDDEN * 2);
  __bf16* qbuf  = (__bf16*)carve((size_t)BATCH * NHEADS * SEQ * DHEAD * 2); // 8 MB
  __bf16* kbuf  = (__bf16*)carve((size_t)BATCH * NHEADS * SEQ * DHEAD * 2);
  __bf16* vtbuf = (__bf16*)carve((size_t)BATCH * NHEADS * SEQ * DHEAD * 2);
  __bf16* attnb = (__bf16*)carve((size_t)MTOT * HIDDEN * 2);

  // 1) fp32 -> bf16 conversions
  {
    int n = MTOT * HIDDEN;
    cvt_f32_bf16<<<(n + 255) / 256, 256, 0, stream>>>(x, xb, n);
    int nw = HIDDEN * HIDDEN;
    cvt_f32_bf16<<<(nw + 255) / 256, 256, 0, stream>>>(Wq, wqb, nw);
    cvt_f32_bf16<<<(nw + 255) / 256, 256, 0, stream>>>(Wk, wkb, nw);
    cvt_f32_bf16<<<(nw + 255) / 256, 256, 0, stream>>>(Wv, wvb, nw);
    cvt_f32_bf16<<<(nw + 255) / 256, 256, 0, stream>>>(Wo, wob, nw);
  }

  // 2) Q/K/V projections (V stored transposed for the PV GEMM)
  dim3 gproj(MTOT / 128, HIDDEN / 64);  // (32, 16)
  proj_qkv<<<gproj, 128, 0, stream>>>(xb, wqb, bq, qbuf, 0);
  proj_qkv<<<gproj, 128, 0, stream>>>(xb, wkb, bk, kbuf, 0);
  proj_qkv<<<gproj, 128, 0, stream>>>(xb, wvb, bv, vtbuf, 1);

  // 3) flash attention, one block row = 128 query rows, 8 waves
  flash_attn<<<dim3(BATCH * NHEADS, SEQ / 128), 256, 0, stream>>>(
      qbuf, kbuf, vtbuf, attnb);

  // 4) output projection (fp32)
  proj_out<<<gproj, 128, 0, stream>>>(attnb, wob, bo, out);
}
